// LocalAttention_12120397709577
// MI455X (gfx1250) — compile-verified, matrix operand order checked by reading, not run on hardware
//
#include <hip/hip_runtime.h>
#include <hip/hip_bf16.h>

// ---------------------------------------------------------------------------
// Local (banded) attention, B=2 N=2048 D=1024 H=16 DH=64, window |i-j|<=64.
// bf16 WMMA (v_wmma_f32_16x16x32_bf16) everywhere; GEMM staging uses
// CDNA5 async LDS DMA (global_load_async_to_lds_b128), double-buffered,
// with a branch-free steady-state pipeline (last k-step peeled).
// ---------------------------------------------------------------------------

typedef __attribute__((ext_vector_type(16))) __bf16 v16bf;
typedef __attribute__((ext_vector_type(8)))  float  v8f;
typedef __attribute__((ext_vector_type(4)))  float  f32x4;   // trivial 16B chunk

union Frag {
    v16bf  f;
    f32x4  q[2];
    __bf16 e[16];
};

static constexpr int NSEQ = 2048;
static constexpr int DMODEL = 1024;
static constexpr int NHEAD = 16;
static constexpr int DHEAD = 64;
static constexpr int MROWS = 2 * NSEQ;            // 4096 total tokens

// GEMM tiling
static constexpr int BM = 128, BN = 128, BK = 32;
static constexpr int LDS_STRIDE = 40;             // padded (16B aligned rows)

// ---------------------------------------------------------------------------
// fp32 -> bf16 conversion (plain, and transposing for weights)
// ---------------------------------------------------------------------------
__global__ void f32_to_bf16_kernel(const float* __restrict__ in,
                                   __bf16* __restrict__ out, int n) {
    int i = blockIdx.x * blockDim.x + threadIdx.x;
    if (i < n) out[i] = (__bf16)in[i];
}

// in[K][Ncols] f32 -> out[Ncols][K] bf16  (one-time cost, tiny vs GEMM time)
__global__ void f32_to_bf16_transpose_kernel(const float* __restrict__ in,
                                             __bf16* __restrict__ out,
                                             int K, int Ncols) {
    int i = blockIdx.x * blockDim.x + threadIdx.x;
    if (i < K * Ncols) {
        int k = i / Ncols, n = i - k * Ncols;
        out[(size_t)n * K + k] = (__bf16)in[i];
    }
}

// ---------------------------------------------------------------------------
// Async LDS staging helper: 16B per call, DMA global -> LDS, no VGPR bounce.
// ---------------------------------------------------------------------------
__device__ __forceinline__ void async_copy_b128(const __bf16* gptr,
                                                const __bf16* lptr) {
    unsigned lds_off = (unsigned)(uintptr_t)lptr;   // low 32b of flat = LDS off
    asm volatile("global_load_async_to_lds_b128 %0, %1, off"
                 :: "v"(lds_off), "v"(gptr) : "memory");
}

// ---------------------------------------------------------------------------
// Block GEMM core: acc[4][2] per wave (64x32 region).
// A[M,K] row-major bf16; Bt[N,K] row-major bf16 (pre-transposed weights).
// 256 threads = 8 waves (2x4 M x N).  Double-buffered async staging with the
// last k-step peeled so the hot loop has constant wait immediates, no branch.
// ---------------------------------------------------------------------------
template <int K>
__device__ __forceinline__ void gemm_block_bf16(
    const __bf16* __restrict__ A, const __bf16* __restrict__ Bt,
    int bm0, int bn0,
    __bf16 (*As)[BM * LDS_STRIDE], __bf16 (*Bs)[BM * LDS_STRIDE],
    v8f acc[4][2])
{
    const int tid     = threadIdx.x;
    const int lane    = tid & 31;
    const int wave    = tid >> 5;
    const int waveM   = wave >> 2;       // 0..1
    const int waveN   = wave & 3;        // 0..3
    const int lmod    = lane & 15;
    const int halfSel = lane >> 4;
    const int kbase   = halfSel * 8;

    // each thread stages 2 chunks of A and 2 chunks of B per k-step
    const int rA = tid >> 2;             // 0..63
    const int cA = tid & 3;

    auto stage = [&](int kk, int buf) {
        #pragma unroll
        for (int it = 0; it < 2; ++it) {
            int r = rA + it * 64;
            async_copy_b128(A + (size_t)(bm0 + r) * K + kk + cA * 8,
                            &As[buf][r * LDS_STRIDE + cA * 8]);
        }
        #pragma unroll
        for (int it = 0; it < 2; ++it) {
            int r = rA + it * 64;
            async_copy_b128(Bt + (size_t)(bn0 + r) * K + kk + cA * 8,
                            &Bs[buf][r * LDS_STRIDE + cA * 8]);
        }
    };

    auto compute = [&](int buf) {
        Frag bfrag[2];
        #pragma unroll
        for (int nt = 0; nt < 2; ++nt) {
            const __bf16* p =
                &Bs[buf][(waveN * 32 + nt * 16 + lmod) * LDS_STRIDE + kbase];
            bfrag[nt].q[0] = *reinterpret_cast<const f32x4*>(p);
            bfrag[nt].q[1] = *reinterpret_cast<const f32x4*>(p + 16);
        }
        #pragma unroll
        for (int mt = 0; mt < 4; ++mt) {
            Frag af;
            const __bf16* p =
                &As[buf][(waveM * 64 + mt * 16 + lmod) * LDS_STRIDE + kbase];
            af.q[0] = *reinterpret_cast<const f32x4*>(p);
            af.q[1] = *reinterpret_cast<const f32x4*>(p + 16);
            #pragma unroll
            for (int nt = 0; nt < 2; ++nt) {
                acc[mt][nt] = __builtin_amdgcn_wmma_f32_16x16x32_bf16(
                    false, af.f, false, bfrag[nt].f, (short)0, acc[mt][nt],
                    false, false);
            }
        }
    };

    stage(0, 0);
    // steady state: always stage next tile, wait only for the previous one
    #pragma unroll 1
    for (int kk = 0; kk < K - BK; kk += BK) {
        const int buf = (kk >> 5) & 1;
        stage(kk + BK, buf ^ 1);
        asm volatile("s_wait_asynccnt 0x4" ::: "memory");
        __syncthreads();
        compute(buf);
        __syncthreads();   // all waves done with buf before it is re-staged
    }
    // peeled last k-step
    asm volatile("s_wait_asynccnt 0x0" ::: "memory");
    __syncthreads();
    compute(((K - BK) >> 5) & 1);
}

// ---------------------------------------------------------------------------
// Kernel 1: QKV projection. xh[4096,1024] @ WqkvT + bqkv; epilogue scatters
// into q[B,H,N,64] (pre-scaled by 1/8), k[B,H,N,64], vT[B,H,64,N].
// ---------------------------------------------------------------------------
__global__ __launch_bounds__(256) void gemm_qkv_kernel(
    const __bf16* __restrict__ A, const __bf16* __restrict__ Bt,
    const float* __restrict__ bias,
    __bf16* __restrict__ qb, __bf16* __restrict__ kb, __bf16* __restrict__ vtb)
{
    __shared__ __bf16 As[2][BM * LDS_STRIDE];
    __shared__ __bf16 Bs[2][BM * LDS_STRIDE];

    const int bm0 = blockIdx.y * BM;
    const int bn0 = blockIdx.x * BN;
    v8f acc[4][2] = {};
    gemm_block_bf16<DMODEL>(A, Bt, bm0, bn0, As, Bs, acc);

    const int lane    = threadIdx.x & 31;
    const int wave    = threadIdx.x >> 5;
    const int waveM   = wave >> 2;
    const int waveN   = wave & 3;
    const int lmod    = lane & 15;
    const int halfSel = lane >> 4;

    #pragma unroll
    for (int mt = 0; mt < 4; ++mt) {
        #pragma unroll
        for (int nt = 0; nt < 2; ++nt) {
            #pragma unroll
            for (int i = 0; i < 8; ++i) {
                int gm = bm0 + waveM * 64 + mt * 16 + i + 8 * halfSel;
                int gn = bn0 + waveN * 32 + nt * 16 + lmod;
                float val = acc[mt][nt][i] + bias[gn];
                int which = gn >> 10;        // 0=q 1=k 2=v
                int dcol  = gn & 1023;
                int hh    = dcol >> 6;
                int dh    = dcol & 63;
                int bb    = gm >> 11;        // token -> batch
                int nn    = gm & 2047;
                size_t bh = (size_t)(bb * NHEAD + hh);
                if (which == 0)
                    qb[(bh * NSEQ + nn) * DHEAD + dh] = (__bf16)(val * 0.125f);
                else if (which == 1)
                    kb[(bh * NSEQ + nn) * DHEAD + dh] = (__bf16)val;
                else
                    vtb[(bh * DHEAD + dh) * NSEQ + nn] = (__bf16)val;
            }
        }
    }
}

// ---------------------------------------------------------------------------
// Kernel 2: banded attention. One wave per 16-query tile (4 waves / block).
// Window of a 16-aligned tile = exactly 9 key tiles (144 keys).
// ---------------------------------------------------------------------------
static constexpr int S_STRIDE = 148;   // f32, padded
static constexpr int P_STRIDE = 160;   // bf16, padded to 5 k-steps of 32

__global__ __launch_bounds__(128) void attn_kernel(
    const __bf16* __restrict__ qb, const __bf16* __restrict__ kb,
    const __bf16* __restrict__ vtb, __bf16* __restrict__ ao)
{
    __shared__ float  Sld[4][16 * S_STRIDE];
    __shared__ __bf16 Pld[4][16 * P_STRIDE];

    const int tid     = threadIdx.x;
    const int wave    = tid >> 5;
    const int lane    = tid & 31;
    const int lmod    = lane & 15;
    const int halfSel = lane >> 4;

    const int gtile = blockIdx.x * 4 + wave;   // 0..4095
    const int bh    = gtile >> 7;              // 128 q-tiles per (b,h)
    const int qt    = gtile & 127;
    const int i0    = qt * 16;

    const __bf16* Q  = qb  + (size_t)bh * NSEQ * DHEAD;
    const __bf16* Kp = kb  + (size_t)bh * NSEQ * DHEAD;
    const __bf16* Vt = vtb + (size_t)bh * DHEAD * NSEQ;
    float*  S = &Sld[wave][0];
    __bf16* P = &Pld[wave][0];

    // Q fragments (rows i0..i0+15, dh 0..63 -> 2 WMMA k-steps)
    Frag aq[2];
    #pragma unroll
    for (int ks = 0; ks < 2; ++ks) {
        const __bf16* p = Q + (size_t)(i0 + lmod) * DHEAD + ks * 32 + halfSel * 8;
        aq[ks].q[0] = *reinterpret_cast<const f32x4*>(p);
        aq[ks].q[1] = *reinterpret_cast<const f32x4*>(p + 16);
    }

    // ---- scores: 9 key tiles x 2 WMMA each ----
    #pragma unroll
    for (int jt = 0; jt < 9; ++jt) {
        const int j0     = i0 - 64 + jt * 16;
        const int keyRow = j0 + lmod;
        const bool inR   = (keyRow >= 0) && (keyRow < NSEQ);
        const int keyCl  = inR ? keyRow : 0;

        v8f accS = {};
        #pragma unroll
        for (int ks = 0; ks < 2; ++ks) {
            Frag bk;
            const __bf16* p = Kp + (size_t)keyCl * DHEAD + ks * 32 + halfSel * 8;
            bk.q[0] = *reinterpret_cast<const f32x4*>(p);
            bk.q[1] = *reinterpret_cast<const f32x4*>(p + 16);
            accS = __builtin_amdgcn_wmma_f32_16x16x32_bf16(
                false, aq[ks].f, false, bk.f, (short)0, accS, false, false);
        }
        #pragma unroll
        for (int i = 0; i < 8; ++i) {
            int m  = i + 8 * halfSel;
            int ig = i0 + m;
            int jg = j0 + lmod;
            bool ok = inR && (jg >= ig - 64) && (jg <= ig + 64);
            S[m * S_STRIDE + jt * 16 + lmod] = ok ? accS[i] : -1e30f;
        }
    }
    __syncthreads();

    // ---- masked softmax: lanes 0..15 own one row each ----
    if (halfSel == 0) {
        const int row = lmod;
        float mx = -1e30f;
        for (int j = 0; j < 144; ++j)
            mx = fmaxf(mx, S[row * S_STRIDE + j]);
        float sum = 0.f;
        for (int j = 0; j < 144; ++j) {
            float e = __expf(S[row * S_STRIDE + j] - mx);
            S[row * S_STRIDE + j] = e;
            sum += e;
        }
        float inv = 1.0f / sum;
        for (int j = 0; j < 144; ++j)
            P[row * P_STRIDE + j] = (__bf16)(S[row * S_STRIDE + j] * inv);
    } else {
        const int row = lmod;                  // zero-pad cols 144..159
        for (int j = 144; j < 160; ++j)
            P[row * P_STRIDE + j] = (__bf16)0.f;
    }
    __syncthreads();

    // ---- O = P[16x160] @ Vwin[160x64] : 5 k-steps x 4 dh-tiles ----
    v8f accO[4] = {};
    #pragma unroll
    for (int ks = 0; ks < 5; ++ks) {
        Frag ap;
        const __bf16* pp = P + lmod * P_STRIDE + ks * 32 + halfSel * 8;
        ap.q[0] = *reinterpret_cast<const f32x4*>(pp);
        ap.q[1] = *reinterpret_cast<const f32x4*>(pp + 16);
        const int kb0 = i0 - 64 + ks * 32 + halfSel * 8;
        #pragma unroll
        for (int nt = 0; nt < 4; ++nt) {
            const int dh = nt * 16 + lmod;
            const __bf16* vrow = Vt + (size_t)dh * NSEQ;
            Frag bv;
            #pragma unroll
            for (int c = 0; c < 2; ++c) {
                int j0c = kb0 + c * 16;
                if (j0c >= 0 && j0c + 7 < NSEQ) {
                    bv.q[c] = *reinterpret_cast<const f32x4*>(vrow + j0c);
                } else {
                    #pragma unroll
                    for (int e = 0; e < 8; ++e) {
                        int j = j0c + e;
                        j = j < 0 ? 0 : (j > NSEQ - 1 ? NSEQ - 1 : j);
                        bv.e[c * 8 + e] = vrow[j];   // P==0 there, value irrelevant
                    }
                }
            }
            accO[nt] = __builtin_amdgcn_wmma_f32_16x16x32_bf16(
                false, ap.f, false, bv.f, (short)0, accO[nt], false, false);
        }
    }

    // ---- write attention output [B,N,D] bf16 ----
    const int b  = bh >> 4;
    const int hh = bh & 15;
    #pragma unroll
    for (int nt = 0; nt < 4; ++nt) {
        #pragma unroll
        for (int i = 0; i < 8; ++i) {
            int m   = i + 8 * halfSel;
            int ig  = i0 + m;
            int col = hh * DHEAD + nt * 16 + lmod;
            ao[((size_t)b * NSEQ + ig) * DMODEL + col] = (__bf16)accO[nt][i];
        }
    }
}

// ---------------------------------------------------------------------------
// Kernel 3: output projection. ao[4096,1024] @ WoutT + bout -> f32
// ---------------------------------------------------------------------------
__global__ __launch_bounds__(256) void gemm_out_kernel(
    const __bf16* __restrict__ A, const __bf16* __restrict__ Bt,
    const float* __restrict__ bias, float* __restrict__ out)
{
    __shared__ __bf16 As[2][BM * LDS_STRIDE];
    __shared__ __bf16 Bs[2][BM * LDS_STRIDE];

    const int bm0 = blockIdx.y * BM;
    const int bn0 = blockIdx.x * BN;
    v8f acc[4][2] = {};
    gemm_block_bf16<DMODEL>(A, Bt, bm0, bn0, As, Bs, acc);

    const int lane    = threadIdx.x & 31;
    const int wave    = threadIdx.x >> 5;
    const int waveM   = wave >> 2;
    const int waveN   = wave & 3;
    const int lmod    = lane & 15;
    const int halfSel = lane >> 4;

    #pragma unroll
    for (int mt = 0; mt < 4; ++mt) {
        #pragma unroll
        for (int nt = 0; nt < 2; ++nt) {
            #pragma unroll
            for (int i = 0; i < 8; ++i) {
                int gm = bm0 + waveM * 64 + mt * 16 + i + 8 * halfSel;
                int gn = bn0 + waveN * 32 + nt * 16 + lmod;
                out[(size_t)gm * DMODEL + gn] = acc[mt][nt][i] + bias[gn];
            }
        }
    }
}

// ---------------------------------------------------------------------------
extern "C" void kernel_launch(void* const* d_in, const int* in_sizes, int n_in,
                              void* d_out, int out_size, void* d_ws, size_t ws_size,
                              hipStream_t stream) {
    const float* x    = (const float*)d_in[0];
    const float* Wqkv = (const float*)d_in[1];
    const float* bqkv = (const float*)d_in[2];
    const float* Wout = (const float*)d_in[3];
    const float* bout = (const float*)d_in[4];
    float* out = (float*)d_out;

    char* ws = (char*)d_ws;
    __bf16* xh    = (__bf16*)ws; ws += (size_t)MROWS * DMODEL * 2;       // 8 MB
    __bf16* wqkvt = (__bf16*)ws; ws += (size_t)DMODEL * 3 * DMODEL * 2;  // 6 MB (transposed [3072][1024])
    __bf16* woutt = (__bf16*)ws; ws += (size_t)DMODEL * DMODEL * 2;      // 2 MB (transposed [1024][1024])
    __bf16* qb    = (__bf16*)ws; ws += (size_t)MROWS * DMODEL * 2;       // 8 MB
    __bf16* kb    = (__bf16*)ws; ws += (size_t)MROWS * DMODEL * 2;       // 8 MB
    __bf16* vtb   = (__bf16*)ws; ws += (size_t)MROWS * DMODEL * 2;       // 8 MB
    __bf16* aob   = (__bf16*)ws; ws += (size_t)MROWS * DMODEL * 2;       // 8 MB
    (void)ws_size;  // total ~48 MB

    const int n_x = MROWS * DMODEL;
    const int n_w = DMODEL * 3 * DMODEL;
    const int n_o = DMODEL * DMODEL;
    f32_to_bf16_kernel<<<(n_x + 255) / 256, 256, 0, stream>>>(x, xh, n_x);
    f32_to_bf16_transpose_kernel<<<(n_w + 255) / 256, 256, 0, stream>>>(
        Wqkv, wqkvt, DMODEL, 3 * DMODEL);
    f32_to_bf16_transpose_kernel<<<(n_o + 255) / 256, 256, 0, stream>>>(
        Wout, woutt, DMODEL, DMODEL);

    gemm_qkv_kernel<<<dim3(3 * DMODEL / BN, MROWS / BM), 256, 0, stream>>>(
        xh, wqkvt, bqkv, qb, kb, vtb);

    attn_kernel<<<(2 * NHEAD * (NSEQ / 16)) / 4, 128, 0, stream>>>(
        qb, kb, vtb, aob);

    gemm_out_kernel<<<dim3(DMODEL / BN, MROWS / BM), 256, 0, stream>>>(
        aob, woutt, bout, out);
}
